// FeatureFusion_77154792506111
// MI455X (gfx1250) — matrix operand order repaired and checked by reading, not compiled, not run on hardware
//
#include <hip/hip_runtime.h>
#include <hip/hip_bf16.h>

// ---------------------------------------------------------------------------
// FeatureFusion for MI455X (gfx1250, wave32, WMMA)
// N=2, L=4096 (64x64), C=256. conf GEMM done in bf16 WMMA (16x16x32),
// rest is bandwidth-bound elementwise / reduction work that lives in L2.
// ---------------------------------------------------------------------------

#define N_B   2
#define L_DIM 4096
#define C_DIM 256
#define HWD   64
#define NCH   16      // column-stat row chunks
#define CHUNK 256     // rows per chunk (NCH*CHUNK == L_DIM)
#define THR_F 0.2f
#define EPS_F 1e-5f

typedef __attribute__((ext_vector_type(16))) __bf16 bf16x16;
typedef __attribute__((ext_vector_type(8)))  float  f32x8;
typedef __attribute__((ext_vector_type(4)))  unsigned int u32x4;

union Frag16 { bf16x16 v; u32x4 q[2]; };

__device__ __forceinline__ unsigned short f32_to_bf16(float f) {
  unsigned int u = __float_as_uint(f);
  u += 0x7FFFu + ((u >> 16) & 1u);           // round-to-nearest-even
  return (unsigned short)(u >> 16);
}

// WGP-scope prefetch (scope field 0): pulls into all cache levels incl. L0.
// __builtin_prefetch can only reach SE/SYS scope, which skips the WGP caches.
__device__ __forceinline__ void prefetch_wgp(const void* p) {
  asm volatile("global_prefetch_b8 %0, off" :: "v"(p));
}

// --------------------------- 1. fp32 -> bf16 -------------------------------
__global__ void cvt_bf16_k(const float* __restrict__ f0, const float* __restrict__ f1,
                           unsigned short* __restrict__ b0, unsigned short* __restrict__ b1) {
  size_t i = (size_t)blockIdx.x * 256 + threadIdx.x;
  b0[i] = f32_to_bf16(f0[i]);
  b1[i] = f32_to_bf16(f1[i]);
}

// --------------------------- 2. conf GEMM (WMMA) ---------------------------
// conf[n][l][s] = dot(feat0[n,l,:], feat1[n,s,:]) / 16
// Block = 128 threads = 4 waves in 2x2; wave tile = 64 rows x 32 cols
// (4x2 tiles of 16x16), K=256 in 8 steps of 32 (unroll 2).
// __launch_bounds__(128,1): allow a big VGPR allocation -> no scratch spills.
__global__ void __launch_bounds__(128, 1)
conf_gemm_k(const unsigned short* __restrict__ A,
            const unsigned short* __restrict__ B,
            float* __restrict__ conf) {
  const int lane = threadIdx.x & 31;
  const int wave = threadIdx.x >> 5;
  const int n    = blockIdx.z;
  const int row0 = blockIdx.y * 128 + (wave >> 1) * 64;
  const int col0 = blockIdx.x * 64  + (wave & 1)  * 32;
  const int r    = lane & 15;
  const int half = lane >> 4;

  const unsigned short* a_base = A + (size_t)n * L_DIM * C_DIM;
  const unsigned short* b_base = B + (size_t)n * L_DIM * C_DIM;

  // Per-lane row pointers are K-invariant: compute once.
  // A: lane<16 -> row r, K chunks {k..k+7, k+16..k+23}; lane>=16 -> +8 shift.
  // B: lane<16 -> col r, K chunks {k..k+7, k+16..k+23}; lane>=16 -> +8 shift.
  const unsigned short* ap[4];
  const unsigned short* bp[2];
#pragma unroll
  for (int i = 0; i < 4; ++i)
    ap[i] = a_base + (size_t)(row0 + i * 16 + r) * C_DIM + half * 8;
#pragma unroll
  for (int j = 0; j < 2; ++j)
    bp[j] = b_base + (size_t)(col0 + j * 16 + r) * C_DIM + half * 8;

  f32x8 acc[4][2];
#pragma unroll
  for (int i = 0; i < 4; ++i)
#pragma unroll
    for (int j = 0; j < 2; ++j)
#pragma unroll
      for (int e = 0; e < 8; ++e) acc[i][j][e] = 0.0f;

#pragma unroll 2
  for (int k = 0; k < C_DIM; k += 32) {
    Frag16 af[4], bfr[2];
#pragma unroll
    for (int i = 0; i < 4; ++i) {
      af[i].q[0] = *(const u32x4*)(ap[i] + k);
      af[i].q[1] = *(const u32x4*)(ap[i] + k + 16);
    }
#pragma unroll
    for (int j = 0; j < 2; ++j) {
      bfr[j].q[0] = *(const u32x4*)(bp[j] + k);
      bfr[j].q[1] = *(const u32x4*)(bp[j] + k + 16);
    }
    if (k + 64 < C_DIM) {                    // warm L0/WGP$ for the next K-slab
      prefetch_wgp(ap[0] + k + 64);
      prefetch_wgp(bp[0] + k + 64);
    }
#pragma unroll
    for (int i = 0; i < 4; ++i)
#pragma unroll
      for (int j = 0; j < 2; ++j)
        acc[i][j] = __builtin_amdgcn_wmma_f32_16x16x32_bf16(
            false, af[i].v, false, bfr[j].v, (short)0, acc[i][j], false, false);
  }

  const float scale = 1.0f / 16.0f;  // 1/sqrt(C)
  float* cb = conf + (size_t)n * L_DIM * L_DIM;
#pragma unroll
  for (int i = 0; i < 4; ++i)
#pragma unroll
    for (int j = 0; j < 2; ++j) {
      const int cr = row0 + i * 16 + half * 8;
      const int cc = col0 + j * 16 + r;
#pragma unroll
      for (int e = 0; e < 8; ++e)
        cb[(size_t)(cr + e) * L_DIM + cc] = acc[i][j][e] * scale;
    }
}

// ------------------- 3. row softmax stats (axis=2: over s) -----------------
__global__ void rowstat_k(const float* __restrict__ conf,
                          float* __restrict__ rmax, float* __restrict__ rsum) {
  const int l = blockIdx.x, n = blockIdx.y, t = threadIdx.x;
  const float* row = conf + ((size_t)n * L_DIM + l) * L_DIM;
  __shared__ float red[256];
  float m = -1e30f;
  for (int s = t; s < L_DIM; s += 256) m = fmaxf(m, row[s]);
  red[t] = m; __syncthreads();
  for (int o = 128; o > 0; o >>= 1) { if (t < o) red[t] = fmaxf(red[t], red[t + o]); __syncthreads(); }
  const float rm = red[0]; __syncthreads();
  float s_ = 0.0f;
  for (int s = t; s < L_DIM; s += 256) s_ += __expf(row[s] - rm);
  red[t] = s_; __syncthreads();
  for (int o = 128; o > 0; o >>= 1) { if (t < o) red[t] += red[t + o]; __syncthreads(); }
  if (t == 0) { rmax[n * L_DIM + l] = rm; rsum[n * L_DIM + l] = red[0]; }
}

// ------------------- 4. col softmax stats (axis=1: over l), chunked --------
__global__ void colmax_part_k(const float* __restrict__ conf, float* __restrict__ part) {
  const int col = blockIdx.x * 256 + threadIdx.x, ch = blockIdx.y, n = blockIdx.z;
  float m = -1e30f;
  for (int r = ch * CHUNK; r < (ch + 1) * CHUNK; ++r)
    m = fmaxf(m, conf[((size_t)n * L_DIM + r) * L_DIM + col]);
  part[((size_t)n * NCH + ch) * L_DIM + col] = m;
}
__global__ void colmax_red_k(const float* __restrict__ part, float* __restrict__ cmax) {
  const int id = blockIdx.x * 256 + threadIdx.x;          // n*L + col
  const int n = id / L_DIM, col = id % L_DIM;
  float m = -1e30f;
  for (int ch = 0; ch < NCH; ++ch) m = fmaxf(m, part[((size_t)n * NCH + ch) * L_DIM + col]);
  cmax[id] = m;
}
__global__ void colsum_part_k(const float* __restrict__ conf, const float* __restrict__ cmax,
                              float* __restrict__ part) {
  const int col = blockIdx.x * 256 + threadIdx.x, ch = blockIdx.y, n = blockIdx.z;
  const float cm = cmax[n * L_DIM + col];
  float s = 0.0f;
  for (int r = ch * CHUNK; r < (ch + 1) * CHUNK; ++r)
    s += __expf(conf[((size_t)n * L_DIM + r) * L_DIM + col] - cm);
  part[((size_t)n * NCH + ch) * L_DIM + col] = s;
}
__global__ void colsum_red_k(const float* __restrict__ part, float* __restrict__ csum) {
  const int id = blockIdx.x * 256 + threadIdx.x;
  const int n = id / L_DIM, col = id % L_DIM;
  float s = 0.0f;
  for (int ch = 0; ch < NCH; ++ch) s += part[((size_t)n * NCH + ch) * L_DIM + col];
  csum[id] = s;
}

// --------- 5. argmax of conf2 = rowSoftmax*colSoftmax along each row -------
__global__ void rowarg2_k(const float* __restrict__ conf,
                          const float* __restrict__ rmax, const float* __restrict__ rsum,
                          const float* __restrict__ cmax, const float* __restrict__ csum,
                          float* __restrict__ rc_val, int* __restrict__ rc_idx) {
  const int l = blockIdx.x, n = blockIdx.y, t = threadIdx.x;
  const float* row = conf + ((size_t)n * L_DIM + l) * L_DIM;
  const float rm = rmax[n * L_DIM + l], ri = 1.0f / rsum[n * L_DIM + l];
  float best = -1.0f; int bi = 0;
  for (int s = t; s < L_DIM; s += 256) {
    const float c = row[s];
    const float v = __expf(c - rm) * ri * __expf(c - cmax[n * L_DIM + s]) / csum[n * L_DIM + s];
    if (v > best) { best = v; bi = s; }
  }
  __shared__ float rv[256]; __shared__ int rix[256];
  rv[t] = best; rix[t] = bi; __syncthreads();
  for (int o = 128; o > 0; o >>= 1) {
    if (t < o && rv[t + o] > rv[t]) { rv[t] = rv[t + o]; rix[t] = rix[t + o]; }
    __syncthreads();
  }
  if (t == 0) { rc_val[n * L_DIM + l] = rv[0]; rc_idx[n * L_DIM + l] = rix[0]; }
}

// --------- 6. argmax of conf2 along each column (chunked + reduce) ---------
__global__ void colarg2_part_k(const float* __restrict__ conf,
                               const float* __restrict__ rmax, const float* __restrict__ rsum,
                               const float* __restrict__ cmax, const float* __restrict__ csum,
                               float* __restrict__ pval, int* __restrict__ pidx) {
  const int col = blockIdx.x * 256 + threadIdx.x, ch = blockIdx.y, n = blockIdx.z;
  const float cm = cmax[n * L_DIM + col], ci = 1.0f / csum[n * L_DIM + col];
  float best = -1.0f; int bi = 0;
  for (int r = ch * CHUNK; r < (ch + 1) * CHUNK; ++r) {
    const float c = conf[((size_t)n * L_DIM + r) * L_DIM + col];
    const float v = __expf(c - rmax[n * L_DIM + r]) / rsum[n * L_DIM + r] * __expf(c - cm) * ci;
    if (v > best) { best = v; bi = r; }
  }
  const size_t o = ((size_t)n * NCH + ch) * L_DIM + col;
  pval[o] = best; pidx[o] = bi;
}
__global__ void colarg2_red_k(const float* __restrict__ pval, const int* __restrict__ pidx,
                              float* __restrict__ cc_val, int* __restrict__ cc_idx) {
  const int id = blockIdx.x * 256 + threadIdx.x;
  const int n = id / L_DIM, col = id % L_DIM;
  float best = -1.0f; int bi = 0;
  for (int ch = 0; ch < NCH; ++ch) {
    const size_t o = ((size_t)n * NCH + ch) * L_DIM + col;
    if (pval[o] > best) { best = pval[o]; bi = pidx[o]; }
  }
  cc_val[id] = best; cc_idx[id] = bi;
}

// --------------------- 7. mutual nearest-neighbor match --------------------
__global__ void match_k(const float* __restrict__ rc_val, const int* __restrict__ rc_idx,
                        const float* __restrict__ cc_val, const int* __restrict__ cc_idx,
                        int* __restrict__ match0, int* __restrict__ matchT) {
  const int id = blockIdx.x * 256 + threadIdx.x;     // n*L + l
  const int n = id / L_DIM, l = id % L_DIM;
  const int s0 = rc_idx[id];
  match0[id] = (rc_val[id] > THR_F && cc_idx[n * L_DIM + s0] == l) ? s0 : -1;
  const int s1 = cc_idx[id];
  matchT[id] = (cc_val[id] > THR_F && rc_idx[n * L_DIM + s1] == l) ? s1 : -1;
}

// ----------------- 8. unfold3x3 + sem + fea (raw-reshape aliasing) ---------
// u[n,l,w9,c] = flat[c*4096 + (h+dy-1)*64 + (w+dx-1)]  (0 outside the image)
__global__ void fea_k(const float* __restrict__ f0, const float* __restrict__ f1,
                      float* __restrict__ fea0, float* __restrict__ fea1) {
  const int l = blockIdx.x, n = blockIdx.y, which = blockIdx.z, t = threadIdx.x;
  const float* src = (which ? f1 : f0) + (size_t)n * L_DIM * C_DIM;
  float* dst = (which ? fea1 : fea0) + ((size_t)n * L_DIM + l) * C_DIM;
  const int h = l >> 6, w = l & 63;
  const float fv = src[(size_t)l * C_DIM + t];
  float u[9];
#pragma unroll
  for (int dy = 0; dy < 3; ++dy)
#pragma unroll
    for (int dx = 0; dx < 3; ++dx) {
      const int y = h + dy - 1, x = w + dx - 1;
      u[dy * 3 + dx] = (y >= 0 && y < HWD && x >= 0 && x < HWD)
                           ? src[(size_t)t * L_DIM + y * HWD + x] : 0.0f;
    }
  __shared__ float red[256]; __shared__ float sem[9];
  for (int k = 0; k < 9; ++k) {
    red[t] = fv * u[k]; __syncthreads();
    for (int o = 128; o > 0; o >>= 1) { if (t < o) red[t] += red[t + o]; __syncthreads(); }
    if (t == 0) sem[k] = red[0] * (1.0f / C_DIM);
    __syncthreads();
  }
  float acc = 0.0f;
#pragma unroll
  for (int k = 0; k < 9; ++k) acc += u[k] * sem[k];
  dst[t] = acc;
}

// ----------------------- 9. sparse gather of s0 / s1 -----------------------
__global__ void gather_k(const float* __restrict__ fea0, const float* __restrict__ fea1,
                         const int* __restrict__ match0, const int* __restrict__ matchT,
                         float* __restrict__ s0b, float* __restrict__ s1b) {
  const int l = blockIdx.x, n = blockIdx.y, t = threadIdx.x;
  const size_t o = ((size_t)n * L_DIM + l) * C_DIM + t;
  const int m0 = match0[n * L_DIM + l];
  s0b[o] = (m0 >= 0) ? fea1[((size_t)n * L_DIM + m0) * C_DIM + t] * (1.0f / L_DIM) : 0.0f;
  const int m1 = matchT[n * L_DIM + l];
  s1b[o] = (m1 >= 0) ? fea0[((size_t)n * L_DIM + m1) * C_DIM + t] * (1.0f / L_DIM) : 0.0f;
}

// ------------------ 10. adaptive pool (160/96) + concat + LN ---------------
__global__ void pool_ln_k(const float* __restrict__ f0, const float* __restrict__ f1,
                          const float* __restrict__ s0b, const float* __restrict__ s1b,
                          const float* __restrict__ lnw, const float* __restrict__ lnb,
                          float* __restrict__ out) {
  const int l = blockIdx.x, row = blockIdx.y, t = threadIdx.x;   // row in [0, 2N)
  const float* base; const float* sbase;
  if (row < N_B) { base = f0 + ((size_t)row * L_DIM + l) * C_DIM;
                   sbase = s0b + ((size_t)row * L_DIM + l) * C_DIM; }
  else           { const int n = row - N_B;
                   base = f1 + ((size_t)n * L_DIM + l) * C_DIM;
                   sbase = s1b + ((size_t)n * L_DIM + l) * C_DIM; }
  float val;
  if (t < 160) {
    const int s = (t * 256) / 160, e = ((t + 1) * 256 + 159) / 160;
    float a = 0.0f; for (int i = s; i < e; ++i) a += base[i];
    val = a / (float)(e - s);
  } else {
    const int j = t - 160;
    const int s = (j * 256) / 96, e = ((j + 1) * 256 + 95) / 96;
    float a = 0.0f; for (int i = s; i < e; ++i) a += sbase[i];
    val = a / (float)(e - s);
  }
  __shared__ float red[256];
  red[t] = val; __syncthreads();
  for (int o = 128; o > 0; o >>= 1) { if (t < o) red[t] += red[t + o]; __syncthreads(); }
  const float mean = red[0] * (1.0f / 256.0f); __syncthreads();
  red[t] = (val - mean) * (val - mean); __syncthreads();
  for (int o = 128; o > 0; o >>= 1) { if (t < o) red[t] += red[t + o]; __syncthreads(); }
  const float var = red[0] * (1.0f / 256.0f);
  out[((size_t)row * L_DIM + l) * C_DIM + t] =
      (val - mean) * rsqrtf(var + EPS_F) * lnw[t] + lnb[t];
}

// ---------------------------------------------------------------------------
extern "C" void kernel_launch(void* const* d_in, const int* in_sizes, int n_in,
                              void* d_out, int out_size, void* d_ws, size_t ws_size,
                              hipStream_t stream) {
  const float* f0  = (const float*)d_in[0];
  const float* f1  = (const float*)d_in[1];
  const float* lnw = (const float*)d_in[2];
  const float* lnb = (const float*)d_in[3];
  float* out = (float*)d_out;

  // workspace carve-out
  char* cur = (char*)d_ws;
  auto alloc = [&](size_t bytes) { char* p = cur; cur += (bytes + 255) & ~(size_t)255; return p; };
  const size_t NLC = (size_t)N_B * L_DIM * C_DIM;
  const size_t NL  = (size_t)N_B * L_DIM;

  unsigned short* bA   = (unsigned short*)alloc(NLC * 2);
  unsigned short* bB   = (unsigned short*)alloc(NLC * 2);
  float* conf  = (float*)alloc((size_t)N_B * L_DIM * L_DIM * 4);
  float* rmax  = (float*)alloc(NL * 4);
  float* rsum  = (float*)alloc(NL * 4);
  float* cmax  = (float*)alloc(NL * 4);
  float* csum  = (float*)alloc(NL * 4);
  float* partF = (float*)alloc(NL * NCH * 4);
  int*   partI = (int*)  alloc(NL * NCH * 4);
  float* rcV   = (float*)alloc(NL * 4);
  int*   rcI   = (int*)  alloc(NL * 4);
  float* ccV   = (float*)alloc(NL * 4);
  int*   ccI   = (int*)  alloc(NL * 4);
  int*   m0    = (int*)  alloc(NL * 4);
  int*   mT    = (int*)  alloc(NL * 4);
  float* fea0  = (float*)alloc(NLC * 4);
  float* fea1  = (float*)alloc(NLC * 4);
  float* s0b   = (float*)alloc(NLC * 4);
  float* s1b   = (float*)alloc(NLC * 4);

  // 1. bf16 convert
  cvt_bf16_k<<<dim3(NLC / 256), dim3(256), 0, stream>>>(f0, f1, bA, bB);
  // 2. conf GEMM (WMMA bf16)
  conf_gemm_k<<<dim3(L_DIM / 64, L_DIM / 128, N_B), dim3(128), 0, stream>>>(bA, bB, conf);
  // 3. row softmax stats
  rowstat_k<<<dim3(L_DIM, N_B), dim3(256), 0, stream>>>(conf, rmax, rsum);
  // 4. col softmax stats
  colmax_part_k<<<dim3(L_DIM / 256, NCH, N_B), dim3(256), 0, stream>>>(conf, partF);
  colmax_red_k<<<dim3(NL / 256), dim3(256), 0, stream>>>(partF, cmax);
  colsum_part_k<<<dim3(L_DIM / 256, NCH, N_B), dim3(256), 0, stream>>>(conf, cmax, partF);
  colsum_red_k<<<dim3(NL / 256), dim3(256), 0, stream>>>(partF, csum);
  // 5-6. argmax of softmax product along rows / columns
  rowarg2_k<<<dim3(L_DIM, N_B), dim3(256), 0, stream>>>(conf, rmax, rsum, cmax, csum, rcV, rcI);
  colarg2_part_k<<<dim3(L_DIM / 256, NCH, N_B), dim3(256), 0, stream>>>(conf, rmax, rsum, cmax, csum, partF, partI);
  colarg2_red_k<<<dim3(NL / 256), dim3(256), 0, stream>>>(partF, partI, ccV, ccI);
  // 7. mutual NN match
  match_k<<<dim3(NL / 256), dim3(256), 0, stream>>>(rcV, rcI, ccV, ccI, m0, mT);
  // 8. unfold + sem + fea
  fea_k<<<dim3(L_DIM, N_B, 2), dim3(256), 0, stream>>>(f0, f1, fea0, fea1);
  // 9. gather s0/s1
  gather_k<<<dim3(L_DIM, N_B), dim3(256), 0, stream>>>(fea0, fea1, m0, mT, s0b, s1b);
  // 10. pool + concat + layernorm -> output (f[:N] then f[N:] == f flat)
  pool_ln_k<<<dim3(L_DIM, 2 * N_B), dim3(256), 0, stream>>>(f0, f1, s0b, s1b, lnw, lnb, out);
}
